// SparseLinear_12249246728781
// MI455X (gfx1250) — matrix operand order repaired and checked by reading, not compiled, not run on hardware
//
#include <hip/hip_runtime.h>
#include <stdint.h>

// Y[4096,4096] = X[4096,4096] * W[4096,4096]^T + bias   (reference: fp32 linear)
//
// Compute-bound GEMM (AI ~683 FLOP/B). bf16x3 fp32-emulation:
//   x = hi(bf16) + lo(bf16 residual);  x*w ~= hi*hi + hi*lo + lo*hi
// accumulated in fp32 by V_WMMA_F32_16X16X32_BF16 (8192 MAC/instr vs 1024 for
// the native f32 WMMA => ~2.7x effective fp32 rate).
//
// Preferred path: one-time split pre-pass into d_ws (memory-bound, ~11us at
// 23.3 TB/s), then a pure-WMMA main loop fed by double-buffered
// global_load_async_to_lds_b128 (ASYNCcnt). Fallback (small d_ws): fused
// in-kernel conversion.

typedef __attribute__((ext_vector_type(16))) __bf16 v16bf;
typedef __attribute__((ext_vector_type(8)))  __bf16 v8bf;
typedef __attribute__((ext_vector_type(8)))  float  v8f;
typedef __attribute__((ext_vector_type(4)))  float  v4f;

#define M_DIM 4096
#define N_DIM 4096
#define K_DIM 4096
#define BM 128
#define BN 128
#define KT 32
#define LDB 40   // bf16 LDS row stride: 80B = 16B-aligned, conflict-free (gcd(20,64)=4, period 16)

#define WMMA_BF16(A, B, C) \
    __builtin_amdgcn_wmma_f32_16x16x32_bf16(false, (A), false, (B), (short)0, (C), false, false)

__device__ __forceinline__ uint32_t lds_off(const void* p) {
    return (uint32_t)(uintptr_t)p;   // low 32 bits of flat shared addr = LDS byte offset
}

__device__ __forceinline__ void async_ld_b128(uint32_t lds_byte, const void* g) {
    uint64_t ga = (uint64_t)(uintptr_t)g;
    asm volatile("global_load_async_to_lds_b128 %0, %1, off"
                 :: "v"(lds_byte), "v"(ga) : "memory");
}

__device__ __forceinline__ __bf16 bf16_hi(float v) { return (__bf16)v; }
__device__ __forceinline__ __bf16 bf16_lo(float v, __bf16 h) {
    return (__bf16)(v - (float)h);
}

// ---------------------------------------------------------------------------
// Pre-pass: split fp32 matrix into bf16 hi / bf16 lo arrays (8 elems/thread)
// ---------------------------------------------------------------------------
__global__ __launch_bounds__(256)
void split_bf16_kernel(const float* __restrict__ src,
                       __bf16* __restrict__ hi, __bf16* __restrict__ lo) {
    const size_t idx = ((size_t)blockIdx.x * 256 + threadIdx.x) * 8;
    v4f a0 = *(const v4f*)(src + idx);
    v4f a1 = *(const v4f*)(src + idx + 4);
    v8bf h, l;
    #pragma unroll
    for (int e = 0; e < 4; ++e) {
        __bf16 h0 = bf16_hi(a0[e]);
        h[e]     = h0;
        l[e]     = bf16_lo(a0[e], h0);
        __bf16 h1 = bf16_hi(a1[e]);
        h[4 + e] = h1;
        l[4 + e] = bf16_lo(a1[e], h1);
    }
    *(v8bf*)(hi + idx) = h;
    *(v8bf*)(lo + idx) = l;
}

// ---------------------------------------------------------------------------
// Main GEMM: pre-split inputs, async global->LDS double-buffered pipeline
// ---------------------------------------------------------------------------
__global__ __launch_bounds__(256)
void gemm_bf16x3_pre(const __bf16* __restrict__ Xhi, const __bf16* __restrict__ Xlo,
                     const __bf16* __restrict__ Whi, const __bf16* __restrict__ Wlo,
                     const float* __restrict__ bias, float* __restrict__ Y) {
    __shared__ __bf16 Ahi[2][BM * LDB], Alo[2][BM * LDB];
    __shared__ __bf16 Bhi[2][BN * LDB], Blo[2][BN * LDB];

    const int tid  = threadIdx.x;
    const int lane = tid & 31;
    const int wave = tid >> 5;
    const int wm   = (wave >> 2) * 64;   // wave M base: 0 / 64
    const int wn   = (wave & 3) * 32;    // wave N base: 0..96
    const int half = lane >> 4;
    const int r    = lane & 15;

    const int mBlock = (int)blockIdx.y * BM;
    const int nBlock = (int)blockIdx.x * BN;

    // Tile = 128 rows x 32 bf16 = 128 x 4 b128-slots; 512 slots/tile, 4 tiles,
    // 256 threads -> 2 slots per thread per tile -> 8 async loads per prefetch.
    auto prefetch = [&](int buf, int kb) {
        const int kBase = kb * KT;
        #pragma unroll
        for (int it = 0; it < 2; ++it) {
            const int s   = it * 256 + tid;
            const int row = s >> 2;          // 0..127
            const int c8  = (s & 3) * 8;     // bf16 element offset: 0,8,16,24
            const size_t gx = (size_t)(mBlock + row) * K_DIM + kBase + c8;
            const size_t gw = (size_t)(nBlock + row) * K_DIM + kBase + c8;
            const int    lo_off = row * LDB + c8;          // 16B-aligned (x2B)
            async_ld_b128(lds_off(&Ahi[buf][lo_off]), Xhi + gx);
            async_ld_b128(lds_off(&Alo[buf][lo_off]), Xlo + gx);
            async_ld_b128(lds_off(&Bhi[buf][lo_off]), Whi + gw);
            async_ld_b128(lds_off(&Blo[buf][lo_off]), Wlo + gw);
        }
    };

    // A fragment, 16-bit A 16x32 layout:
    //   lanes 0-15: K 0-7 (v0-3), K 16-23 (v4-7); lanes 16-31: K 8-15, K 24-31
    auto load_afrag = [&](const __bf16* base, int i) -> v16bf {
        const __bf16* p = base + (wm + i * 16 + r) * LDB;
        v8bf p0 = *(const v8bf*)(p + half * 8);
        v8bf p1 = *(const v8bf*)(p + 16 + half * 8);
        return __builtin_shufflevector(p0, p1, 0, 1, 2, 3, 4, 5, 6, 7,
                                                8, 9, 10, 11, 12, 13, 14, 15);
    };
    // B fragment, 32x16 layout: N = lane&15; lanes 0-15 K 0-7, K 16-23; lanes 16-31 K 8-15, K 24-31
    auto load_bfrag = [&](const __bf16* base, int j) -> v16bf {
        const __bf16* p = base + (wn + j * 16 + r) * LDB + 8 * half;
        v8bf p0 = *(const v8bf*)(p);
        v8bf p1 = *(const v8bf*)(p + 16);
        return __builtin_shufflevector(p0, p1, 0, 1, 2, 3, 4, 5, 6, 7,
                                                8, 9, 10, 11, 12, 13, 14, 15);
    };

    v8f acc[4][2];
    #pragma unroll
    for (int i = 0; i < 4; ++i)
        #pragma unroll
        for (int j = 0; j < 2; ++j)
            acc[i][j] = (v8f){0.f, 0.f, 0.f, 0.f, 0.f, 0.f, 0.f, 0.f};

    prefetch(0, 0);

    const int nK = K_DIM / KT;   // 128
    for (int kb = 0; kb < nK; ++kb) {
        const int cur = kb & 1;
        if (kb + 1 < nK) {
            prefetch(cur ^ 1, kb + 1);
            // 8 just issued for next buffer; async loads retire in order, so
            // waiting to <=8 guarantees the current buffer's 8 are done.
            asm volatile("s_wait_asynccnt 8" ::: "memory");
        } else {
            asm volatile("s_wait_asynccnt 0" ::: "memory");
        }
        __syncthreads();   // all waves' tiles for `cur` visible

        v16bf bhf[2], blf[2];
        #pragma unroll
        for (int j = 0; j < 2; ++j) {
            bhf[j] = load_bfrag(&Bhi[cur][0], j);
            blf[j] = load_bfrag(&Blo[cur][0], j);
        }
        #pragma unroll
        for (int i = 0; i < 4; ++i) {
            v16bf ahf = load_afrag(&Ahi[cur][0], i);
            v16bf alf = load_afrag(&Alo[cur][0], i);
            #pragma unroll
            for (int j = 0; j < 2; ++j) {
                v8f c = acc[i][j];
                c = WMMA_BF16(alf, bhf[j], c);   // smallest terms first
                c = WMMA_BF16(ahf, blf[j], c);
                c = WMMA_BF16(ahf, bhf[j], c);
                acc[i][j] = c;
            }
        }
        __syncthreads();   // everyone done reading `cur` before it is refilled
    }

    // epilogue: C/D layout (VGPR v -> row v + 8*half, lane&15 -> col), fused bias
    #pragma unroll
    for (int j = 0; j < 2; ++j) {
        const int col = nBlock + wn + j * 16 + r;
        const float bv = bias[col];
        #pragma unroll
        for (int i = 0; i < 4; ++i) {
            #pragma unroll
            for (int v = 0; v < 8; ++v) {
                const int row = mBlock + wm + i * 16 + v + 8 * half;
                Y[(size_t)row * N_DIM + col] = acc[i][j][v] + bv;
            }
        }
    }
}

// ---------------------------------------------------------------------------
// Fallback: fused in-kernel conversion (used when d_ws is too small)
// ---------------------------------------------------------------------------
__global__ __launch_bounds__(256)
void gemm_bf16x3_fused(const float* __restrict__ X, const float* __restrict__ W,
                       const float* __restrict__ bias, float* __restrict__ Y) {
    __shared__ __bf16 Ahi[BM * LDB], Alo[BM * LDB];
    __shared__ __bf16 Bhi[BN * LDB], Blo[BN * LDB];

    const int tid  = threadIdx.x;
    const int lane = tid & 31;
    const int wave = tid >> 5;
    const int wm   = (wave >> 2) * 64;
    const int wn   = (wave & 3) * 32;
    const int half = lane >> 4;
    const int r    = lane & 15;

    const int mBlock = (int)blockIdx.y * BM;
    const int nBlock = (int)blockIdx.x * BN;

    const int srow = tid >> 1;
    const int kOff = (tid & 1) * 16;

    const float* gA = X + (size_t)(mBlock + srow) * K_DIM + kOff;
    const float* gB = W + (size_t)(nBlock + srow) * K_DIM + kOff;

    float ra[16], rb[16];
    auto load_raw = [&](int kb) {
        const float* pa = gA + kb * KT;
        const float* pb = gB + kb * KT;
        #pragma unroll
        for (int q = 0; q < 4; ++q) {
            v4f va = *(const v4f*)(pa + q * 4);
            v4f vb = *(const v4f*)(pb + q * 4);
            #pragma unroll
            for (int e = 0; e < 4; ++e) { ra[q * 4 + e] = va[e]; rb[q * 4 + e] = vb[e]; }
        }
    };
    auto split_store = [&]() {
        v8bf ah[2], al[2], bh[2], bl[2];
        #pragma unroll
        for (int g = 0; g < 2; ++g) {
            #pragma unroll
            for (int e = 0; e < 8; ++e) {
                float va = ra[g * 8 + e];
                __bf16 ha = bf16_hi(va);
                ah[g][e] = ha;
                al[g][e] = bf16_lo(va, ha);
                float vb = rb[g * 8 + e];
                __bf16 hb = bf16_hi(vb);
                bh[g][e] = hb;
                bl[g][e] = bf16_lo(vb, hb);
            }
        }
        #pragma unroll
        for (int g = 0; g < 2; ++g) {
            *(v8bf*)&Ahi[srow * LDB + kOff + g * 8] = ah[g];
            *(v8bf*)&Alo[srow * LDB + kOff + g * 8] = al[g];
            *(v8bf*)&Bhi[srow * LDB + kOff + g * 8] = bh[g];
            *(v8bf*)&Blo[srow * LDB + kOff + g * 8] = bl[g];
        }
    };
    auto load_afrag = [&](const __bf16* base, int i) -> v16bf {
        const __bf16* p = base + (wm + i * 16 + r) * LDB;
        v8bf p0 = *(const v8bf*)(p + half * 8);
        v8bf p1 = *(const v8bf*)(p + 16 + half * 8);
        return __builtin_shufflevector(p0, p1, 0, 1, 2, 3, 4, 5, 6, 7,
                                                8, 9, 10, 11, 12, 13, 14, 15);
    };
    auto load_bfrag = [&](const __bf16* base, int j) -> v16bf {
        const __bf16* p = base + (wn + j * 16 + r) * LDB + 8 * half;
        v8bf p0 = *(const v8bf*)(p);
        v8bf p1 = *(const v8bf*)(p + 16);
        return __builtin_shufflevector(p0, p1, 0, 1, 2, 3, 4, 5, 6, 7,
                                                8, 9, 10, 11, 12, 13, 14, 15);
    };

    v8f acc[4][2];
    #pragma unroll
    for (int i = 0; i < 4; ++i)
        #pragma unroll
        for (int j = 0; j < 2; ++j)
            acc[i][j] = (v8f){0.f, 0.f, 0.f, 0.f, 0.f, 0.f, 0.f, 0.f};

    load_raw(0);
    const int nK = K_DIM / KT;
    for (int kb = 0; kb < nK; ++kb) {
        split_store();
        __syncthreads();
        if (kb + 1 < nK) load_raw(kb + 1);

        v16bf bhf[2], blf[2];
        #pragma unroll
        for (int j = 0; j < 2; ++j) { bhf[j] = load_bfrag(Bhi, j); blf[j] = load_bfrag(Blo, j); }
        #pragma unroll
        for (int i = 0; i < 4; ++i) {
            v16bf ahf = load_afrag(Ahi, i);
            v16bf alf = load_afrag(Alo, i);
            #pragma unroll
            for (int j = 0; j < 2; ++j) {
                v8f c = acc[i][j];
                c = WMMA_BF16(alf, bhf[j], c);
                c = WMMA_BF16(ahf, blf[j], c);
                c = WMMA_BF16(ahf, bhf[j], c);
                acc[i][j] = c;
            }
        }
        __syncthreads();
    }

    #pragma unroll
    for (int j = 0; j < 2; ++j) {
        const int col = nBlock + wn + j * 16 + r;
        const float bv = bias[col];
        #pragma unroll
        for (int i = 0; i < 4; ++i)
            #pragma unroll
            for (int v = 0; v < 8; ++v) {
                const int row = mBlock + wm + i * 16 + v + 8 * half;
                Y[(size_t)row * N_DIM + col] = acc[i][j][v] + bv;
            }
    }
}

extern "C" void kernel_launch(void* const* d_in, const int* in_sizes, int n_in,
                              void* d_out, int out_size, void* d_ws, size_t ws_size,
                              hipStream_t stream) {
    const float* X    = (const float*)d_in[0];
    const float* W    = (const float*)d_in[1];
    const float* bias = (const float*)d_in[2];
    float* Y          = (float*)d_out;

    const size_t elemsX = (size_t)M_DIM * K_DIM;
    const size_t elemsW = (size_t)N_DIM * K_DIM;
    const size_t need   = (elemsX + elemsW) * 2 * sizeof(uint16_t);  // hi+lo, both mats

    dim3 grid(N_DIM / BN, M_DIM / BM);
    dim3 block(256);

    if (ws_size >= need) {
        __bf16* Xhi = (__bf16*)d_ws;
        __bf16* Xlo = Xhi + elemsX;
        __bf16* Whi = Xlo + elemsX;
        __bf16* Wlo = Whi + elemsW;
        const int nbX = (int)(elemsX / 8 / 256);   // 8192 blocks
        const int nbW = (int)(elemsW / 8 / 256);
        hipLaunchKernelGGL(split_bf16_kernel, dim3(nbX), block, 0, stream, X, Xhi, Xlo);
        hipLaunchKernelGGL(split_bf16_kernel, dim3(nbW), block, 0, stream, W, Whi, Wlo);
        hipLaunchKernelGGL(gemm_bf16x3_pre, grid, block, 0, stream,
                           Xhi, Xlo, Whi, Wlo, bias, Y);
    } else {
        hipLaunchKernelGGL(gemm_bf16x3_fused, grid, block, 0, stream, X, W, bias, Y);
    }
}
